// SelectiveMoEFusionBlock_231928234756
// MI455X (gfx1250) — matrix operand order repaired and checked by reading, not compiled, not run on hardware
//
#include <hip/hip_runtime.h>
#include <hip/hip_bf16.h>
#include <cstdint>

// ---------- types ----------
typedef __bf16 bf16_t;
typedef bf16_t v16bf __attribute__((ext_vector_type(16)));
typedef bf16_t v8bf  __attribute__((ext_vector_type(8)));
typedef bf16_t v4bf  __attribute__((ext_vector_type(4)));
typedef float  v8f   __attribute__((ext_vector_type(8)));

// ---------- problem constants ----------
#define NPTS   200000
#define NCH    128
#define NHID   64
#define NHF    116
#define NWF    200
#define NTILES (NPTS / 16)          // 12500 exactly (no tail)

// ---------- d_ws / LDS weight layout (bytes) ----------
#define OFF_FG1   0                 // 128x64  bf16 frags: 16 frags * 1024B
#define OFF_RT1   16384             // 128x64  bf16 frags
#define OFF_LE1   32768             // 256x128 bf16 frags: 64 frags
#define OFF_LE2   98304             // 128x128 bf16 frags: 32 frags
#define OFF_SMALL 131072            // f32 region
#define SMALL_F32 1280
#define WS_BYTES  (OFF_SMALL + SMALL_F32 * 4)   // 136192 (multiple of 16)

// indices into the f32 "smalls" region
#define S_FGB1 0
#define S_FGW2 64
#define S_FGB2 128
#define S_RTB1 129
#define S_RTW2 193       // 64x3 row-major
#define S_RTB2 385
#define S_LEB1 388
#define S_LEB2 516
#define S_LNG  644
#define S_LNB  772
#define S_RTX  900       // rt_w1 rows 128..130 (3 x 64)

// ---------- per-wave LDS scratch ----------
// 8K bufA + 4K bufX + 4K bufS + 1K meta + 768B qbuf
#define WAVE_SCRATCH 18176
#define LDS_TOTAL (WS_BYTES + 8 * WAVE_SCRATCH)   // 281600 < 320KB WGP LDS

#define LDS_FENCE()   asm volatile("s_wait_dscnt 0x0" ::: "memory")
#define ASYNC_FENCE() asm volatile("s_wait_asynccnt 0x0" ::: "memory")

// Async DMA 16 bytes global -> LDS (CDNA5 GLOBAL_LOAD_ASYNC_TO_LDS_B128,
// GV mode: 64-bit VGPR address, LDS byte offset in the dest VGPR).
static __device__ __forceinline__ void async_g2l_b128(uint32_t lds_off,
                                                      const void* gptr) {
    asm volatile("global_load_async_to_lds_b128 %0, %1, off"
                 :: "v"(lds_off), "v"((uint64_t)(uintptr_t)gptr)
                 : "memory");
}

static __device__ __forceinline__ float clamp01(float v) {
    return fminf(fmaxf(v, 0.f), 1.f);
}

// ---------- WMMA fragment loads ----------
// A (16x32 bf16) from a row-major [16][K] bf16 LDS buffer.
// lane<16 : M=lane,    K = kt*32 + {0..7, 16..23}
// lane>=16: M=lane-16, K = kt*32 + {8..15, 24..31}
static __device__ __forceinline__ v16bf load_a_frag(const bf16_t* act, int K,
                                                    int kt, int lane) {
    const int half = lane >> 4;
    const int m    = lane & 15;
    const bf16_t* rp = act + m * K + kt * 32 + (half ? 8 : 0);
    v8bf lo = *(const v8bf*)(rp);
    v8bf hi = *(const v8bf*)(rp + 16);
    return __builtin_shufflevector(lo, hi, 0, 1, 2, 3, 4, 5, 6, 7,
                                   8, 9, 10, 11, 12, 13, 14, 15);
}

// B (32x16 bf16) fragment pre-permuted by the prep kernel: 32B per lane.
static __device__ __forceinline__ v16bf load_b_frag(const bf16_t* w, int frag,
                                                    int lane) {
    return *(const v16bf*)(w + frag * 512 + lane * 16);
}

// ---------- prep kernel: f32 weights -> bf16 WMMA-B-fragment layout ----------
static __device__ __forceinline__ void pack_one(const float* __restrict__ W,
                                                int Nout, int NT,
                                                bf16_t* __restrict__ out, int e) {
    const int frag = e >> 9, within = e & 511;
    const int l = within >> 4, j = within & 15;
    const int kt = frag / NT, nt = frag % NT;
    const int half = l >> 4;
    const int n = nt * 16 + (l & 15);
    const int k = kt * 32 + half * 8 + (j & 7) + (j >> 3) * 16;
    out[e] = (bf16_t)W[k * Nout + n];
}

__global__ void moe_prep(const float* __restrict__ fg_w1, const float* __restrict__ fg_b1,
                         const float* __restrict__ fg_w2, const float* __restrict__ fg_b2,
                         const float* __restrict__ rt_w1, const float* __restrict__ rt_b1,
                         const float* __restrict__ rt_w2, const float* __restrict__ rt_b2,
                         const float* __restrict__ le_w1, const float* __restrict__ le_b1,
                         const float* __restrict__ le_w2, const float* __restrict__ le_b2,
                         const float* __restrict__ ln_g,  const float* __restrict__ ln_b,
                         char* __restrict__ ws) {
    const int tid    = blockIdx.x * blockDim.x + threadIdx.x;
    const int stride = gridDim.x * blockDim.x;
    bf16_t* fg1 = (bf16_t*)(ws + OFF_FG1);
    bf16_t* rt1 = (bf16_t*)(ws + OFF_RT1);
    bf16_t* le1 = (bf16_t*)(ws + OFF_LE1);
    bf16_t* le2 = (bf16_t*)(ws + OFF_LE2);
    for (int e = tid; e < 8192;  e += stride) pack_one(fg_w1, 64,  4, fg1, e);
    for (int e = tid; e < 8192;  e += stride) pack_one(rt_w1, 64,  4, rt1, e);
    for (int e = tid; e < 32768; e += stride) pack_one(le_w1, 128, 8, le1, e);
    for (int e = tid; e < 16384; e += stride) pack_one(le_w2, 128, 8, le2, e);
    float* sm = (float*)(ws + OFF_SMALL);
    for (int i = tid; i < SMALL_F32; i += stride) {
        float v = 0.f;
        if      (i < 64)   v = fg_b1[i];
        else if (i < 128)  v = fg_w2[i - 64];
        else if (i == 128) v = fg_b2[0];
        else if (i < 193)  v = rt_b1[i - 129];
        else if (i < 385)  v = rt_w2[i - 193];
        else if (i < 388)  v = rt_b2[i - 385];
        else if (i < 516)  v = le_b1[i - 388];
        else if (i < 644)  v = le_b2[i - 516];
        else if (i < 772)  v = ln_g[i - 644];
        else if (i < 900)  v = ln_b[i - 772];
        else if (i < 1092) { int r = (i - 900) >> 6, n = (i - 900) & 63;
                             v = rt_w1[(128 + r) * 64 + n]; }
        sm[i] = v;
    }
}

// ---------- fused main kernel ----------
__global__ __launch_bounds__(256) void moe_fused(
    const float* __restrict__ x, const int* __restrict__ indices,
    const float* __restrict__ vs, const float* __restrict__ pcr,
    const float* __restrict__ trans, const float* __restrict__ img,
    const int* __restrict__ p_imh, const int* __restrict__ p_imw,
    const char* __restrict__ ws, float* __restrict__ out) {
    extern __shared__ char smem[];
    const int tid  = threadIdx.x;
    const int lane = tid & 31;
    const int wave = tid >> 5;
    const int half = lane >> 4;
    const int col  = lane & 15;

    // stage all weights from d_ws into LDS once per block
    {
        const int4* src = (const int4*)ws;
        int4* dst = (int4*)smem;
        for (int i = tid; i < WS_BYTES / 16; i += 256) dst[i] = src[i];
    }
    __syncthreads();

    const bf16_t* FG1 = (const bf16_t*)(smem + OFF_FG1);
    const bf16_t* RT1 = (const bf16_t*)(smem + OFF_RT1);
    const bf16_t* LE1 = (const bf16_t*)(smem + OFF_LE1);
    const bf16_t* LE2 = (const bf16_t*)(smem + OFF_LE2);
    const float*  SM  = (const float*)(smem + OFF_SMALL);

    char*   wb     = smem + WS_BYTES + wave * WAVE_SCRATCH;
    float*  bufA   = (float*)wb;                 // act f32 / le1-act bf16 / x f32 (8KB)
    bf16_t* bufX   = (bf16_t*)(wb + 8192);       // x tile bf16 (4KB)
    bf16_t* bufS   = (bf16_t*)(wb + 12288);      // sampled tile bf16 (4KB)
    float*  deltaF = (float*)(wb + 8192);        // delta f32, aliases bufX+bufS
    float*  meta   = (float*)(wb + 16384);       // 16 pts x 16 f32
    float*  qbuf   = (float*)(wb + 17408);       // 176 f32 quality staging

    const int   imh = p_imh[0], imw = p_imw[0];
    const float cvx = vs[0] * 8.f, cvy = vs[1] * 8.f, cvz = vs[2] * 8.f;
    const float pc0 = pcr[0], pc1 = pcr[1], pc2 = pcr[2];
    const int   ihw = NHF * NWF;

    for (int base = blockIdx.x * 8; base < NTILES; base += gridDim.x * 8) {
        const int tile = base + wave;
        if (tile >= NTILES) continue;
        const int p0 = tile * 16;

        __builtin_prefetch(x + (size_t)p0 * NCH + lane * 16, 0, 1);

        // ---- stage x tile (f32 -> bf16, coalesced) ----
        {
            const float4* xs = (const float4*)(x + (size_t)p0 * NCH);
            for (int i = lane; i < 512; i += 32) {
                float4 v = xs[i];
                v4bf b = {(bf16_t)v.x, (bf16_t)v.y, (bf16_t)v.z, (bf16_t)v.w};
                *(v4bf*)(bufX + i * 4) = b;
            }
        }

        // ---- per-point geometry / projection meta (lanes 0..15) ----
        if (lane < 16) {
            const int p = p0 + lane;
            int4 id = *(const int4*)(indices + p * 4);   // b, z, y, x
            float cx = (float)id.w * cvx + pc0 + cvx * 0.5f;
            float cy = (float)id.z * cvy + pc1 + cvy * 0.5f;
            float cz = (float)id.y * cvz + pc2 + cvz * 0.5f;
            float rn = fminf(sqrtf(cx * cx + cy * cy) * (1.f / 72.f), 1.f);
            const float* P = trans + id.x * 12;
            float pu = P[0] * cx + P[1] * cy + P[2]  * cz + P[3];
            float pv = P[4] * cx + P[5] * cy + P[6]  * cz + P[7];
            float pd = P[8] * cx + P[9] * cy + P[10] * cz + P[11];
            float safe = fmaxf(pd, 1e-5f);
            float uf = pu / safe * ((float)NWF / (float)imw);
            float vf = pv / safe * ((float)NHF / (float)imh);
            float un = 2.f * (uf / (float)(NWF - 1)) - 1.f;
            float vn = 2.f * (vf / (float)(NHF - 1)) - 1.f;
            float valid = (pd > 1e-5f && fabsf(un) <= 1.f && fabsf(vn) <= 1.f)
                              ? 1.f : 0.f;
            float* mp = meta + lane * 16;
            mp[0] = floorf(uf); mp[1] = floorf(vf);
            mp[2] = uf - floorf(uf); mp[3] = vf - floorf(vf);
            mp[4] = (float)id.x; mp[5] = valid; mp[6] = rn;
        }
        LDS_FENCE();

        // ---- cache x A-fragments (shared by fg / rt / le1) ----
        v16bf ah[8];
#pragma unroll
        for (int kt = 0; kt < 4; kt++) ah[kt] = load_a_frag(bufX, 128, kt, lane);

        // ---- fg MLP layer 1: x(128) -> relu(64) ----
#pragma unroll
        for (int nt = 0; nt < 4; nt++) {
            v8f acc = {0.f, 0.f, 0.f, 0.f, 0.f, 0.f, 0.f, 0.f};
#pragma unroll
            for (int kt = 0; kt < 4; kt++)
                acc = __builtin_amdgcn_wmma_f32_16x16x32_bf16(
                    false, ah[kt], false, load_b_frag(FG1, kt * 4 + nt, lane),
                    (short)0, acc, false, false);
            const int n = nt * 16 + col;
            const float bias = SM[S_FGB1 + n];
#pragma unroll
            for (int i = 0; i < 8; i++) {
                const int M = i + 8 * half;
                bufA[M * 64 + n] = fmaxf(acc[i] + bias, 0.f);
            }
        }
        LDS_FENCE();
        if (lane < 16) {   // fg layer 2 + sigmoid -> p_fg
            float s = SM[S_FGB2];
            for (int h = 0; h < 64; h++) s += bufA[lane * 64 + h] * SM[S_FGW2 + h];
            meta[lane * 16 + 7] = 1.f / (1.f + __expf(-s));
        }
        LDS_FENCE();

        // ---- route MLP layer 1: [x,range,p_fg,valid](131) -> relu(64) ----
        float rng8[8], pf8[8], vl8[8];
#pragma unroll
        for (int i = 0; i < 8; i++) {
            const int M = i + 8 * half;
            rng8[i] = meta[M * 16 + 6];
            pf8[i]  = meta[M * 16 + 7];
            vl8[i]  = meta[M * 16 + 5];
        }
#pragma unroll
        for (int nt = 0; nt < 4; nt++) {
            v8f acc = {0.f, 0.f, 0.f, 0.f, 0.f, 0.f, 0.f, 0.f};
#pragma unroll
            for (int kt = 0; kt < 4; kt++)
                acc = __builtin_amdgcn_wmma_f32_16x16x32_bf16(
                    false, ah[kt], false, load_b_frag(RT1, kt * 4 + nt, lane),
                    (short)0, acc, false, false);
            const int n = nt * 16 + col;
            const float w0 = SM[S_RTX + n], w1 = SM[S_RTX + 64 + n],
                        w2 = SM[S_RTX + 128 + n], bias = SM[S_RTB1 + n];
#pragma unroll
            for (int i = 0; i < 8; i++) {
                const int M = i + 8 * half;
                float v = acc[i] + rng8[i] * w0 + pf8[i] * w1 + vl8[i] * w2 + bias;
                bufA[M * 64 + n] = fmaxf(v, 0.f);
            }
        }
        LDS_FENCE();
        if (lane < 16) {   // route layer 2 + softmax
            float l0 = SM[S_RTB2], l1 = SM[S_RTB2 + 1], l2 = SM[S_RTB2 + 2];
            for (int h = 0; h < 64; h++) {
                const float a = bufA[lane * 64 + h];
                l0 += a * SM[S_RTW2 + h * 3];
                l1 += a * SM[S_RTW2 + h * 3 + 1];
                l2 += a * SM[S_RTW2 + h * 3 + 2];
            }
            float mx = fmaxf(l0, fmaxf(l1, l2));
            float e0 = __expf(l0 - mx), e1 = __expf(l1 - mx), e2 = __expf(l2 - mx);
            float inv = 1.f / (e0 + e1 + e2);
            meta[lane * 16 + 8]  = e0 * inv;
            meta[lane * 16 + 9]  = e1 * inv;
            meta[lane * 16 + 10] = e2 * inv;
        }
        LDS_FENCE();

        // ---- bilinear gather of 128 channels (skip invalid: delta masked) ----
        for (int m = 0; m < 16; m++) {
            const float* mp = meta + m * 16;
            const float vmask = mp[5];
            if (vmask == 0.f) {
#pragma unroll
                for (int j = 0; j < 4; j++)
                    bufS[m * NCH + j * 32 + lane] = (bf16_t)0.f;
                continue;
            }
            const int x0i = (int)mp[0], y0i = (int)mp[1];
            const float fx = mp[2], fy = mp[3];
            const int bi = (int)mp[4];
            float tw[4]; int toff[4];
#pragma unroll
            for (int t = 0; t < 4; t++) {
                const int dx = t & 1, dy = t >> 1;
                const int xi = x0i + dx, yi = y0i + dy;
                const float w = (dx ? fx : 1.f - fx) * (dy ? fy : 1.f - fy);
                const bool inb = (xi >= 0) & (xi < NWF) & (yi >= 0) & (yi < NHF);
                tw[t]   = inb ? w : 0.f;
                toff[t] = min(max(yi, 0), NHF - 1) * NWF + min(max(xi, 0), NWF - 1);
            }
            const float* ib = img + (size_t)bi * NCH * ihw;
#pragma unroll
            for (int j = 0; j < 4; j++) {
                const int c = j * 32 + lane;
                const float* cb = ib + (size_t)c * ihw;
                float s = tw[0] * cb[toff[0]] + tw[1] * cb[toff[1]] +
                          tw[2] * cb[toff[2]] + tw[3] * cb[toff[3]];
                bufS[m * NCH + c] = (bf16_t)s;
            }
        }
        LDS_FENCE();

        // ---- le MLP layer 1: [x|sampled](256) -> relu(128) ----
#pragma unroll
        for (int kt = 0; kt < 4; kt++)
            ah[4 + kt] = load_a_frag(bufS, 128, kt, lane);
        bf16_t* bufH = (bf16_t*)bufA;
#pragma unroll
        for (int nt = 0; nt < 8; nt++) {
            v8f acc = {0.f, 0.f, 0.f, 0.f, 0.f, 0.f, 0.f, 0.f};
#pragma unroll
            for (int kt = 0; kt < 8; kt++)
                acc = __builtin_amdgcn_wmma_f32_16x16x32_bf16(
                    false, ah[kt], false, load_b_frag(LE1, kt * 8 + nt, lane),
                    (short)0, acc, false, false);
            const int n = nt * 16 + col;
            const float bias = SM[S_LEB1 + n];
#pragma unroll
            for (int i = 0; i < 8; i++) {
                const int M = i + 8 * half;
                bufH[M * 128 + n] = (bf16_t)fmaxf(acc[i] + bias, 0.f);
            }
        }
        LDS_FENCE();

        // ---- le MLP layer 2: (128)->(128), scaled by route[1]*valid ----
        v16bf a2[4];
#pragma unroll
        for (int kt = 0; kt < 4; kt++) a2[kt] = load_a_frag(bufH, 128, kt, lane);
        float sc8[8];
#pragma unroll
        for (int i = 0; i < 8; i++) {
            const int M = i + 8 * half;
            sc8[i] = meta[M * 16 + 9] * meta[M * 16 + 5];
        }
        LDS_FENCE();   // a2 fragments now in VGPRs; bufA (= bufH) is dead

        // Async DMA the x tile (f32) into bufA while the le2 WMMAs run:
        // pure global->LDS copy on the ASYNC path, hidden under matrix math.
        {
            const char* gx = (const char*)(x + (size_t)p0 * NCH);
            const uint32_t lbase = (uint32_t)(uintptr_t)(void*)bufA;
            for (int i = lane; i < 512; i += 32)
                async_g2l_b128(lbase + i * 16, gx + i * 16);
        }

#pragma unroll
        for (int nt = 0; nt < 8; nt++) {
            v8f acc = {0.f, 0.f, 0.f, 0.f, 0.f, 0.f, 0.f, 0.f};
#pragma unroll
            for (int kt = 0; kt < 4; kt++)
                acc = __builtin_amdgcn_wmma_f32_16x16x32_bf16(
                    false, a2[kt], false, load_b_frag(LE2, kt * 8 + nt, lane),
                    (short)0, acc, false, false);
            const int n = nt * 16 + col;
            const float bias = SM[S_LEB2 + n];
#pragma unroll
            for (int i = 0; i < 8; i++) {
                const int M = i + 8 * half;
                deltaF[M * 128 + n] = (acc[i] + bias) * sc8[i];
            }
        }
        ASYNC_FENCE();   // x f32 tile resident in bufA
        LDS_FENCE();     // delta stores complete

        // ---- epilogue phase A: per-point stats via cross-half shuffle ----
        {
            const int m = col;
            const float* xr = (const float*)bufA + m * NCH + half * 64;
            const float* dr = deltaF + m * NCH + half * 64;
            float s1 = 0.f, s2 = 0.f, sd = 0.f, sx = 0.f;
            for (int c = 0; c < 64; c++) {
                const float xv = xr[c], dv = dr[c], h = xv + dv;
                s1 += h; s2 += h * h; sd += dv * dv; sx += xv * xv;
            }
            s1 += __shfl_xor(s1, 16);
            s2 += __shfl_xor(s2, 16);
            sd += __shfl_xor(sd, 16);
            sx += __shfl_xor(sx, 16);
            const float mu   = s1 * (1.f / 128.f);
            const float var  = s2 * (1.f / 128.f) - mu * mu;
            const float rstd = rsqrtf(var + 1e-5f);
            const float gain = clamp01(1.f - __expf(-sqrtf(sd) / (sqrtf(sx) + 1e-6f)));
            if (half == 0) {
                meta[m * 16 + 11] = mu;
                meta[m * 16 + 12] = rstd;
                meta[m * 16 + 13] = gain;
            }
        }
        LDS_FENCE();

        // ---- epilogue phase B: coalesced float4 layernorm stores ----
        {
            float* og = out + (size_t)p0 * NCH;
            const float* xa = (const float*)bufA;
            for (int i = lane; i < 512; i += 32) {
                const int idx = i * 4;
                const int mm  = idx >> 7;
                const int cc  = idx & 127;
                const float mu   = meta[mm * 16 + 11];
                const float rstd = meta[mm * 16 + 12];
                const float4 xv = *(const float4*)(xa + idx);
                const float4 dv = *(const float4*)(deltaF + idx);
                float4 o;
                o.x = ((xv.x + dv.x) - mu) * rstd * SM[S_LNG + cc]     + SM[S_LNB + cc];
                o.y = ((xv.y + dv.y) - mu) * rstd * SM[S_LNG + cc + 1] + SM[S_LNB + cc + 1];
                o.z = ((xv.z + dv.z) - mu) * rstd * SM[S_LNG + cc + 2] + SM[S_LNB + cc + 2];
                o.w = ((xv.w + dv.w) - mu) * rstd * SM[S_LNG + cc + 3] + SM[S_LNB + cc + 3];
                *(float4*)(og + idx) = o;
            }
        }

        // ---- quality: stage 16x11 in LDS, then coalesced float4 stores ----
        if (lane < 16) {
            const float* mp = meta + lane * 16;
            float* q = qbuf + lane * 11;
            q[0] = clamp01(mp[8]);  q[1] = clamp01(mp[9]);
            q[2] = clamp01(mp[10]); q[3] = clamp01(mp[7]);
            q[4] = clamp01(mp[6]);  q[5] = clamp01(mp[5]);
            q[6] = 0.f;             q[7] = mp[13];
            q[8] = 0.f; q[9] = 0.f; q[10] = 0.f;
        }
        LDS_FENCE();
        {
            float* qg = out + (size_t)NPTS * NCH + (size_t)p0 * 11;
            for (int i = lane; i < 44; i += 32)
                *(float4*)(qg + i * 4) = *(const float4*)(qbuf + i * 4);
        }
        LDS_FENCE();
    }
}

// ---------- launch ----------
extern "C" void kernel_launch(void* const* d_in, const int* in_sizes, int n_in,
                              void* d_out, int out_size, void* d_ws, size_t ws_size,
                              hipStream_t stream) {
    const float* x       = (const float*)d_in[0];
    const int*   indices = (const int*)d_in[1];
    const float* vsz     = (const float*)d_in[2];
    const float* pcr     = (const float*)d_in[3];
    const float* trans   = (const float*)d_in[4];
    const float* img     = (const float*)d_in[5];
    const int*   p_imh   = (const int*)d_in[6];
    const int*   p_imw   = (const int*)d_in[7];
    const float* fg_w1 = (const float*)d_in[8];
    const float* fg_b1 = (const float*)d_in[9];
    const float* fg_w2 = (const float*)d_in[10];
    const float* fg_b2 = (const float*)d_in[11];
    const float* rt_w1 = (const float*)d_in[12];
    const float* rt_b1 = (const float*)d_in[13];
    const float* rt_w2 = (const float*)d_in[14];
    const float* rt_b2 = (const float*)d_in[15];
    const float* le_w1 = (const float*)d_in[16];
    const float* le_b1 = (const float*)d_in[17];
    const float* le_w2 = (const float*)d_in[18];
    const float* le_b2 = (const float*)d_in[19];
    const float* ln_g  = (const float*)d_in[20];
    const float* ln_b  = (const float*)d_in[21];
    float* out = (float*)d_out;
    char*  ws  = (char*)d_ws;

    moe_prep<<<64, 256, 0, stream>>>(fg_w1, fg_b1, fg_w2, fg_b2,
                                     rt_w1, rt_b1, rt_w2, rt_b2,
                                     le_w1, le_b1, le_w2, le_b2,
                                     ln_g, ln_b, ws);
    moe_fused<<<320, 256, LDS_TOTAL, stream>>>(x, indices, vsz, pcr, trans, img,
                                               p_imh, p_imw, ws, out);
}
